// KalmanFilter_11441792876570
// MI455X (gfx1250) — matrix operand order repaired and checked by reading, not compiled, not run on hardware
//
#include <hip/hip_runtime.h>

// Problem constants (match reference)
#define T_STEPS 1024
#define BATCH   2048
#define SDIM    32
#define DDIM    16
#define KDIM    (T_STEPS * DDIM)   // 16384

typedef __attribute__((ext_vector_type(16))) _Float16 v16h;
typedef __attribute__((ext_vector_type(8)))  float    v8f;
typedef __attribute__((ext_vector_type(4)))  unsigned uintx4;
typedef __attribute__((ext_vector_type(8)))  int      intx8;
typedef __attribute__((ext_vector_type(4)))  int      intx4;

__device__ __forceinline__ float softplus_f(float x) {
    return log1pf(expf(x));
}

// ---------------------------------------------------------------------------
// Kernel 1: data-independent covariance recursion (1 workgroup, serial in t).
// Produces per-step tables:
//   Lc[t][d][e] = L_t[e][d]  (lane d in kernel 2 reads a contiguous row)
//     where L_t L_tᵀ = S_t⁻¹,  L_t = chol(S_t)^{-T}
//   Mc[t][s][d] = M_t[d][s]  with M_t = S_t⁻¹ H P_t   (mean-gain, Sinv folded)
//   constOut[0] = Σ_t logdet(S_t) + T*D*ln(2π)
// ---------------------------------------------------------------------------
__global__ __launch_bounds__(256)
void kf_cov_kernel(const float* __restrict__ Fg, const float* __restrict__ scr,
                   const float* __restrict__ Hg, const float* __restrict__ ocr,
                   float* __restrict__ Lc, float* __restrict__ Mc,
                   float* __restrict__ constOut)
{
    __shared__ float P [SDIM][SDIM + 1];
    __shared__ float T1[SDIM][SDIM + 1];
    __shared__ float P2[SDIM][SDIM + 1];
    __shared__ float HP[DDIM][SDIM + 1];
    __shared__ float U1[DDIM][SDIM + 1];
    __shared__ float Ml[DDIM][SDIM + 1];
    __shared__ float SC[DDIM][DDIM + 1];
    __shared__ float CI[DDIM][DDIM + 1];

    const int tid = threadIdx.x;
    float ldSum = 0.f;                       // only tid 0's copy is used

    // P = I
    for (int ii = 0; ii < 4; ++ii) {
        int i = tid * 4 + ii, r = i >> 5, c = i & 31;
        P[r][c] = (r == c) ? 1.f : 0.f;
    }
    __syncthreads();

    for (int t = 0; t < T_STEPS; ++t) {
        // T1 = F @ P
        {
            int i0 = tid * 4, r = i0 >> 5, c0 = i0 & 31;
            float s0 = 0.f, s1 = 0.f, s2 = 0.f, s3 = 0.f;
            for (int k = 0; k < SDIM; ++k) {
                float f = Fg[r * SDIM + k];
                s0 += f * P[k][c0 + 0]; s1 += f * P[k][c0 + 1];
                s2 += f * P[k][c0 + 2]; s3 += f * P[k][c0 + 3];
            }
            T1[r][c0 + 0] = s0; T1[r][c0 + 1] = s1;
            T1[r][c0 + 2] = s2; T1[r][c0 + 3] = s3;
        }
        __syncthreads();
        // P2 = T1 @ Fᵀ + diag(softplus(state_cov_raw))
        {
            int i0 = tid * 4, r = i0 >> 5, c0 = i0 & 31;
            for (int ii = 0; ii < 4; ++ii) {
                int c = c0 + ii;
                float s = 0.f;
                for (int k = 0; k < SDIM; ++k) s += T1[r][k] * Fg[c * SDIM + k];
                if (r == c) s += softplus_f(scr[r]);
                P2[r][c] = s;
            }
        }
        __syncthreads();
        // HP = H @ P2     [16,32]
        {
            int i0 = tid * 2;
            for (int ii = 0; ii < 2; ++ii) {
                int i = i0 + ii, d = i >> 5, c = i & 31;
                float s = 0.f;
                for (int k = 0; k < SDIM; ++k) s += Hg[d * SDIM + k] * P2[k][c];
                HP[d][c] = s;
            }
        }
        __syncthreads();
        // S = HP @ Hᵀ + diag(softplus(obs_cov_raw))   [16,16]
        {
            int d = tid >> 4, e = tid & 15;
            float s = 0.f;
            for (int k = 0; k < SDIM; ++k) s += HP[d][k] * Hg[e * SDIM + k];
            if (d == e) s += softplus_f(ocr[d]);
            SC[d][e] = s;
        }
        __syncthreads();
        // In-wave (wave 0, lockstep, DS ops in-order): Cholesky of S (lower,
        // in place), logdet accumulation, CI = chol(S)^{-1} (lower triangular)
        if (tid < 32) {
            const int l = tid;
            for (int j = 0; j < 16; ++j) {
                float dj = sqrtf(SC[j][j]);
                if (l == 0) ldSum += 2.f * logf(dj);
                __builtin_amdgcn_wave_barrier();
                if (l > j && l < 16) SC[l][j] = SC[l][j] / dj;
                if (l == j)          SC[j][j] = dj;
                __builtin_amdgcn_wave_barrier();
                if (l > j && l < 16) {
                    float alj = SC[l][j];
                    for (int c2 = j + 1; c2 <= l; ++c2)
                        SC[l][c2] -= alj * SC[c2][j];
                }
                __builtin_amdgcn_wave_barrier();
            }
            if (l < 16) {                       // column l of CI, lane-local
                for (int i2 = 0; i2 < l; ++i2) CI[i2][l] = 0.f;
                CI[l][l] = 1.f / SC[l][l];
                for (int i2 = l + 1; i2 < 16; ++i2) {
                    float s = 0.f;
                    for (int k2 = l; k2 < i2; ++k2) s += SC[i2][k2] * CI[k2][l];
                    CI[i2][l] = -s / SC[i2][i2];
                }
            }
        }
        __syncthreads();
        // U1 = CI @ HP    [16,32]
        {
            int i0 = tid * 2;
            for (int ii = 0; ii < 2; ++ii) {
                int i = i0 + ii, d = i >> 5, c = i & 31;
                float s = 0.f;
                for (int k = 0; k < DDIM; ++k) s += CI[d][k] * HP[k][c];
                U1[d][c] = s;
            }
        }
        __syncthreads();
        // M = CIᵀ @ U1 = S⁻¹ H P2 ; emit Mc (transposed) and Lc (= CI rows)
        {
            int i0 = tid * 2;
            for (int ii = 0; ii < 2; ++ii) {
                int i = i0 + ii, d = i >> 5, c = i & 31;
                float s = 0.f;
                for (int k = 0; k < DDIM; ++k) s += CI[k][d] * U1[k][c];
                Ml[d][c] = s;
                Mc[((size_t)t * SDIM + c) * DDIM + d] = s;
            }
            int d2 = tid >> 4, e = tid & 15;
            Lc[((size_t)t * DDIM + d2) * DDIM + e] = CI[d2][e];
        }
        __syncthreads();
        // newP = P2 - HPᵀ @ M
        {
            int i0 = tid * 4;
            for (int ii = 0; ii < 4; ++ii) {
                int i = i0 + ii, r = i >> 5, c = i & 31;
                float s = 0.f;
                for (int k = 0; k < DDIM; ++k) s += HP[k][r] * Ml[k][c];
                P[r][c] = P2[r][c] - s;
            }
        }
        __syncthreads();
    }
    if (tid == 0)
        constOut[0] = ldSum + (float)T_STEPS * (float)DDIM * logf(6.283185307179586f);
}

// ---------------------------------------------------------------------------
// Kernel 2: batch mean recursion, one wave per batch element (lane = state
// dim), wave-synchronous LDS broadcasts (no barriers needed within a wave).
// Emits Y[b][t*16+d] = (innov_t[b] @ L_t)[d] in f16 (row-major [B, K]).
// ---------------------------------------------------------------------------
__global__ __launch_bounds__(256)
void kf_batch_kernel(const float* __restrict__ obs,
                     const float* __restrict__ Fg, const float* __restrict__ Hg,
                     const float* __restrict__ Lc, const float* __restrict__ Mc,
                     _Float16* __restrict__ Y)
{
    __shared__ float xch[8][64];
    const int tid  = threadIdx.x;
    const int w    = tid >> 5;
    const int lane = tid & 31;
    const int b    = blockIdx.x * 8 + w;

    // Per-lane constant rows: F row `lane` (for upd), H row `lane` (for innov)
    float Fr[SDIM], Hr[SDIM];
    #pragma unroll
    for (int k4 = 0; k4 < 8; ++k4) {
        float4 fv = reinterpret_cast<const float4*>(Fg + lane * SDIM)[k4];
        Fr[k4*4+0] = fv.x; Fr[k4*4+1] = fv.y; Fr[k4*4+2] = fv.z; Fr[k4*4+3] = fv.w;
        float4 hv = make_float4(0.f, 0.f, 0.f, 0.f);
        if (lane < DDIM)
            hv = reinterpret_cast<const float4*>(Hg + lane * SDIM)[k4];
        Hr[k4*4+0] = hv.x; Hr[k4*4+1] = hv.y; Hr[k4*4+2] = hv.z; Hr[k4*4+3] = hv.w;
    }

    float m = 0.f;   // state_mean[b][lane]
    for (int t = 0; t < T_STEPS; ++t) {
        // upd[lane] = Σ_s m[s] * F[lane][s]
        xch[w][lane] = m;
        __builtin_amdgcn_wave_barrier();
        float upd = 0.f;
        #pragma unroll
        for (int k4 = 0; k4 < 8; ++k4) {
            float4 v = reinterpret_cast<const float4*>(&xch[w][0])[k4];
            upd += v.x * Fr[k4*4+0] + v.y * Fr[k4*4+1]
                 + v.z * Fr[k4*4+2] + v.w * Fr[k4*4+3];
        }
        __builtin_amdgcn_wave_barrier();
        // innov[d] = obs[t][b][d] - Σ_s upd[s] * H[d][s]   (lanes 0..15)
        xch[w][lane] = upd;
        __builtin_amdgcn_wave_barrier();
        if (lane < DDIM) {
            float acc = 0.f;
            #pragma unroll
            for (int k4 = 0; k4 < 8; ++k4) {
                float4 v = reinterpret_cast<const float4*>(&xch[w][0])[k4];
                acc += v.x * Hr[k4*4+0] + v.y * Hr[k4*4+1]
                     + v.z * Hr[k4*4+2] + v.w * Hr[k4*4+3];
            }
            float iv = obs[((size_t)t * BATCH + b) * DDIM + lane] - acc;
            xch[w][32 + lane] = iv;
        }
        __builtin_amdgcn_wave_barrier();
        float in[DDIM];
        #pragma unroll
        for (int k4 = 0; k4 < 4; ++k4) {
            float4 v = reinterpret_cast<const float4*>(&xch[w][32])[k4];
            in[k4*4+0] = v.x; in[k4*4+1] = v.y; in[k4*4+2] = v.z; in[k4*4+3] = v.w;
        }
        // new_m[lane] = upd[lane] + Σ_d innov[d] * M_t[d][lane]
        float mnew = upd;
        {
            const float* Mrow = Mc + ((size_t)t * SDIM + lane) * DDIM;
            #pragma unroll
            for (int k4 = 0; k4 < 4; ++k4) {
                float4 v = reinterpret_cast<const float4*>(Mrow)[k4];
                mnew += in[k4*4+0] * v.x + in[k4*4+1] * v.y
                      + in[k4*4+2] * v.z + in[k4*4+3] * v.w;
            }
        }
        // y[d] = Σ_e innov[e] * L_t[e][d]  -> f16 stream
        if (lane < DDIM) {
            const float* Lrow = Lc + ((size_t)t * DDIM + lane) * DDIM;
            float y = 0.f;
            #pragma unroll
            for (int k4 = 0; k4 < 4; ++k4) {
                float4 v = reinterpret_cast<const float4*>(Lrow)[k4];
                y += in[k4*4+0] * v.x + in[k4*4+1] * v.y
                   + in[k4*4+2] * v.z + in[k4*4+3] * v.w;
            }
            Y[(size_t)b * KDIM + t * DDIM + lane] = (_Float16)y;
        }
        m = mnew;
    }
}

// ---------------------------------------------------------------------------
// Kernel 3: ll = -0.5 * (Y @ Yᵀ + constTerm), 2048x2048xK=16384, f16 WMMA.
// 128x128 block tile, 8 waves (4x2), each wave 32x64 = 2x4 WMMA tiles.
// Tiles staged by the Tensor Data Mover (tensor_load_to_lds), double-buffered,
// with TDM padding reproducing a 72-half LDS row stride (16B-aligned b128,
// low-conflict fragment reads). TENSORcnt tracks completion.
// ---------------------------------------------------------------------------
#define KTILE 64
#define LDSTR 72    // halves per row: 64 data + 8 pad (TDM pad: 4 DW / 32 DW)

// Issue one TDM 2D tile load: 128 rows x KTILE halves from Y into LDS.
__device__ __forceinline__ void tdm_load_tile(unsigned ldsOff,
                                              const _Float16* gptr)
{
    unsigned long long ga = (unsigned long long)(size_t)gptr;
    uintx4 g0;
    g0[0] = 1u;                                        // count=1, user mode
    g0[1] = ldsOff;                                    // lds_addr (bytes)
    g0[2] = (unsigned)(ga & 0xFFFFFFFFull);            // global_addr[31:0]
    g0[3] = (unsigned)((ga >> 32) & 0x01FFFFFFull)     // global_addr[56:32]
          | (2u << 30);                                // type = 2 (image)
    intx8 g1;
    g1[0] = (int)((1u << 16)      // data_size = 1 (2-byte elements)
                | (1u << 20)      // pad_enable
                | (4u << 22)      // pad_interval: 32 DWORDs (=64 halves)
                | (3u << 25));    // pad_amount: 4 DWORDs (=8 halves)
    g1[1] = (int)(((unsigned)KDIM & 0xFFFFu) << 16);   // tensor_dim0 lo16
    g1[2] = (int)((((unsigned)KDIM >> 16) & 0xFFFFu)   // tensor_dim0 hi16
                | (((unsigned)BATCH & 0xFFFFu) << 16));// tensor_dim1 lo16
    g1[3] = (int)((((unsigned)BATCH >> 16) & 0xFFFFu)  // tensor_dim1 hi16
                | ((unsigned)KTILE << 16));            // tile_dim0
    g1[4] = (int)128u;                                 // tile_dim1 (tile_dim2=0)
    g1[5] = (int)(unsigned)KDIM;                       // tensor_dim0_stride lo32
    g1[6] = 0;                                         // stride hi16 / dim1_stride
    g1[7] = 0;
    intx4 gz4 = {0, 0, 0, 0};                          // groups 2/3: unused (2D)
    intx8 gz8 = {0, 0, 0, 0, 0, 0, 0, 0};              // group 4: unused
    __builtin_amdgcn_tensor_load_to_lds(g0, g1, gz4, gz4, gz8, 0);
}

__global__ __launch_bounds__(256)
void kf_gemm_kernel(const _Float16* __restrict__ Y,
                    const float* __restrict__ constOut,
                    float* __restrict__ out)
{
    __shared__ _Float16 sA[2][128 * LDSTR];
    __shared__ _Float16 sB[2][128 * LDSTR];

    const int tid  = threadIdx.x;
    const int lane = tid & 31;
    const int wid  = tid >> 5;
    const int waveRow = wid & 3;      // 4 groups of 32 output rows
    const int waveCol = wid >> 2;     // 2 groups of 64 output cols
    const int lrow = lane & 15;
    const int hi   = lane >> 4;

    const int rowBase = blockIdx.y * 128;
    const int colBase = blockIdx.x * 128;

    const unsigned ldsA[2] = { (unsigned)(size_t)&sA[0][0],
                               (unsigned)(size_t)&sA[1][0] };
    const unsigned ldsB[2] = { (unsigned)(size_t)&sB[0][0],
                               (unsigned)(size_t)&sB[1][0] };

    v8f acc[2][4];
    const v8f vzero = {0.f, 0.f, 0.f, 0.f, 0.f, 0.f, 0.f, 0.f};
    #pragma unroll
    for (int i = 0; i < 2; ++i)
        #pragma unroll
        for (int j = 0; j < 4; ++j) acc[i][j] = vzero;

    const int nChunks = KDIM / KTILE;   // 256

    // Prologue: TDM-load chunk 0 into buffer 0 (wave 0 issues; EXEC ignored).
    if (wid == 0) {
        tdm_load_tile(ldsA[0], Y + (size_t)rowBase * KDIM);
        tdm_load_tile(ldsB[0], Y + (size_t)colBase * KDIM);
    }

    for (int c = 0; c < nChunks; ++c) {
        const int cur = c & 1;
        if (wid == 0) {
            if (c + 1 < nChunks) {
                // Prefetch next chunk into the other buffer, then wait for
                // the current chunk's two TDM ops (in-order per wave).
                tdm_load_tile(ldsA[cur ^ 1],
                              Y + (size_t)rowBase * KDIM + (c + 1) * KTILE);
                tdm_load_tile(ldsB[cur ^ 1],
                              Y + (size_t)colBase * KDIM + (c + 1) * KTILE);
                __builtin_amdgcn_s_wait_tensorcnt((short)2);
            } else {
                __builtin_amdgcn_s_wait_tensorcnt((short)0);
            }
        }
        __syncthreads();   // current tiles visible to all waves

        union FU { v16h v; unsigned u[8]; };
        #pragma unroll
        for (int ks = 0; ks < KTILE; ks += 32) {
            FU a[2], bf[4];
            // A fragment (16-bit A 16x32 layout): lane holds row m = lrow,
            // VGPR v -> K = ks + (v>=4?16:0) + hi*8 + (v&3)*2.
            #pragma unroll
            for (int tm = 0; tm < 2; ++tm) {
                int row = waveRow * 32 + tm * 16 + lrow;
                #pragma unroll
                for (int v = 0; v < 8; ++v) {
                    int k = ks + ((v >= 4) ? 16 : 0) + hi * 8 + (v & 3) * 2;
                    a[tm].u[v] = *reinterpret_cast<const unsigned*>(
                        &sA[cur][row * LDSTR + k]);
                }
            }
            // B fragment (32x16): N = lrow, VGPR v -> K = ks + (v>=4?16:0) + hi*8 + (v&3)*2.
            #pragma unroll
            for (int tn = 0; tn < 4; ++tn) {
                int row = waveCol * 64 + tn * 16 + lrow;
                #pragma unroll
                for (int v = 0; v < 8; ++v) {
                    int k = ks + ((v >= 4) ? 16 : 0) + hi * 8 + (v & 3) * 2;
                    bf[tn].u[v] = *reinterpret_cast<const unsigned*>(
                        &sB[cur][row * LDSTR + k]);
                }
            }
            #pragma unroll
            for (int tm = 0; tm < 2; ++tm)
                #pragma unroll
                for (int tn = 0; tn < 4; ++tn)
                    acc[tm][tn] = __builtin_amdgcn_wmma_f32_16x16x32_f16(
                        false, a[tm].v, false, bf[tn].v,
                        (short)0, acc[tm][tn], false, false);
        }
        __syncthreads();   // all waves done reading before buffer reuse
    }

    const float cterm = constOut[0];
    #pragma unroll
    for (int tm = 0; tm < 2; ++tm) {
        #pragma unroll
        for (int tn = 0; tn < 4; ++tn) {
            int col = colBase + waveCol * 64 + tn * 16 + lrow;
            #pragma unroll
            for (int r = 0; r < 8; ++r) {
                // C layout: VGPR r -> M = r + hi*8, N = lrow
                int rowg = rowBase + waveRow * 32 + tm * 16 + hi * 8 + r;
                out[(size_t)rowg * BATCH + col] = -0.5f * (acc[tm][tn][r] + cterm);
            }
        }
    }
}

// ---------------------------------------------------------------------------
extern "C" void kernel_launch(void* const* d_in, const int* in_sizes, int n_in,
                              void* d_out, int out_size, void* d_ws, size_t ws_size,
                              hipStream_t stream)
{
    const float* obs = (const float*)d_in[0];   // [T,B,16]
    const float* F   = (const float*)d_in[1];   // [32,32]
    const float* scr = (const float*)d_in[2];   // [32]
    const float* H   = (const float*)d_in[3];   // [16,32]
    const float* ocr = (const float*)d_in[4];   // [16]
    float* out = (float*)d_out;                 // [2048,2048]

    char* base = (char*)d_ws;
    float*    constP = (float*)base;                                   // 1 f32
    float*    Lc = (float*)(base + 256);                               // 1 MB
    float*    Mc = (float*)(base + 256 + (size_t)T_STEPS*DDIM*DDIM*4); // 2 MB
    _Float16* Yb = (_Float16*)(base + 256
                               + (size_t)T_STEPS*DDIM*DDIM*4
                               + (size_t)T_STEPS*SDIM*DDIM*4);         // 67 MB

    kf_cov_kernel  <<<1,              256, 0, stream>>>(F, scr, H, ocr, Lc, Mc, constP);
    kf_batch_kernel<<<BATCH / 8,      256, 0, stream>>>(obs, F, H, Lc, Mc, Yb);
    kf_gemm_kernel <<<dim3(16, 16),   256, 0, stream>>>(Yb, constP, out);
}
